// PMPNN_Baseline_CPD_66073776881711
// MI455X (gfx1250) — compile-verified, matrix operand order checked by reading, not run on hardware
//
#include <hip/hip_runtime.h>
#include <hip/hip_bf16.h>
#include <math.h>

#define BSZ     4
#define NRES    1024
#define KNBR    48
#define NV      (BSZ*NRES)      // 4096 nodes
#define MEDGE   (NV*KNBR)       // 196608 edges
#define HDIM    128
#define VOCABSZ 21
#define NLAY    3

typedef _Float16 h4  __attribute__((ext_vector_type(4)));
typedef _Float16 h8  __attribute__((ext_vector_type(8)));
typedef _Float16 h16 __attribute__((ext_vector_type(16)));
typedef float    f4  __attribute__((ext_vector_type(4)));
typedef float    f8  __attribute__((ext_vector_type(8)));

__device__ __forceinline__ float gelu_exact(float x) {
    return 0.5f * x * (1.0f + erff(x * 0.70710678118654752f));
}

// atom order in scratch: 0=Ca 1=N 2=C 3=O 4=Cb
__constant__ unsigned char PAIR_A[24] = {1,2,3,4, 0,0,0,0, 1,1,1, 4,4, 3, 1,2,3,4, 2,3,4, 2,3, 2};
__constant__ unsigned char PAIR_B[24] = {1,2,3,4, 1,2,3,4, 2,3,4, 2,3, 2, 0,0,0,0, 1,1,1, 4,4, 3};

// ---------------------------------------------------------------- geometry
__global__ void geom_k(const float* __restrict__ X, float* __restrict__ atoms) {
    int idx = blockIdx.x * blockDim.x + threadIdx.x;
    if (idx >= NV) return;
    const float* x = X + (size_t)idx * 12;   // [4][3] = N, Ca, C, O
    float Nx = x[0], Ny = x[1],  Nz = x[2];
    float Ax = x[3], Ay = x[4],  Az = x[5];  // Ca
    float Cx = x[6], Cy = x[7],  Cz = x[8];
    float Ox = x[9], Oy = x[10], Oz = x[11];
    float bx = Ax - Nx, by = Ay - Ny, bz = Az - Nz;
    float cx = Cx - Ax, cy = Cy - Ay, cz = Cz - Az;
    float ax = by * cz - bz * cy;
    float ay = bz * cx - bx * cz;
    float az = bx * cy - by * cx;
    float Bx = -0.58273431f*ax + 0.56802827f*bx - 0.54067466f*cx + Ax;
    float By = -0.58273431f*ay + 0.56802827f*by - 0.54067466f*cy + Ay;
    float Bz = -0.58273431f*az + 0.56802827f*bz - 0.54067466f*cz + Az;
    float* o = atoms + (size_t)idx * 15;
    o[0]=Ax; o[1]=Ay; o[2]=Az;     // Ca
    o[3]=Nx; o[4]=Ny; o[5]=Nz;     // N
    o[6]=Cx; o[7]=Cy; o[8]=Cz;     // C
    o[9]=Ox; o[10]=Oy; o[11]=Oz;   // O
    o[12]=Bx; o[13]=By; o[14]=Bz;  // Cb
}

// ------------------------------------------------------------ top-K (48/1024)
__global__ void __launch_bounds__(256) topk_k(const float* __restrict__ atoms,
                                              const float* __restrict__ mask,
                                              int* __restrict__ Eidx,
                                              float* __restrict__ Dn) {
    __shared__ float dist[NRES];
    __shared__ float redf[256];
    __shared__ unsigned long long redk[256];
    const int bn = blockIdx.x;           // 0..NV-1
    const int b  = bn / NRES;
    const int t  = threadIdx.x;
    const float* ci = atoms + (size_t)bn * 15;
    const float cix = ci[0], ciy = ci[1], ciz = ci[2];
    const float mi = mask[bn];
    float m2r[4];
    float localmax = 0.f;
    #pragma unroll
    for (int u = 0; u < 4; ++u) {
        int j = t + u * 256;
        const float* cj = atoms + ((size_t)b * NRES + j) * 15;
        float dx = cix - cj[0], dy = ciy - cj[1], dz = ciz - cj[2];
        float d  = sqrtf(dx*dx + dy*dy + dz*dz + 1e-6f);
        float m2 = mi * mask[b * NRES + j];
        m2r[u] = m2;
        float Dm = m2 * d;
        dist[j] = Dm;
        localmax = fmaxf(localmax, Dm);
    }
    redf[t] = localmax; __syncthreads();
    for (int st = 128; st > 0; st >>= 1) {
        if (t < st) redf[t] = fmaxf(redf[t], redf[t + st]);
        __syncthreads();
    }
    float Dmax = redf[0]; __syncthreads();
    #pragma unroll
    for (int u = 0; u < 4; ++u) {
        int j = t + u * 256;
        dist[j] = dist[j] + (1.f - m2r[u]) * Dmax;
    }
    __syncthreads();
    for (int it = 0; it < KNBR; ++it) {
        unsigned long long best = ~0ull;
        #pragma unroll
        for (int u = 0; u < 4; ++u) {
            int j = t + u * 256;
            unsigned long long key =
                ((unsigned long long)__float_as_uint(dist[j]) << 32) | (unsigned)j;
            best = key < best ? key : best;
        }
        redk[t] = best; __syncthreads();
        for (int st = 128; st > 0; st >>= 1) {
            if (t < st) { unsigned long long o = redk[t + st]; if (o < redk[t]) redk[t] = o; }
            __syncthreads();
        }
        if (t == 0) {
            unsigned long long kb = redk[0];
            int js = (int)(kb & 0xffffffffu);
            Eidx[(size_t)bn * KNBR + it] = js;
            Dn  [(size_t)bn * KNBR + it] = __uint_as_float((unsigned)(kb >> 32));
            dist[js] = 3.0e38f;
        }
        __syncthreads();
    }
}

// ---------------------------------------------------------- mask_attend (enc)
__global__ void maskatt_k(const float* __restrict__ mask, const int* __restrict__ Eidx,
                          float* __restrict__ matt) {
    size_t m = (size_t)blockIdx.x * blockDim.x + threadIdx.x;
    if (m >= (size_t)MEDGE) return;
    int bn = (int)(m / KNBR);
    int b  = bn / NRES;
    matt[m] = mask[bn] * mask[b * NRES + Eidx[m]];
}

// -------------------------------------------------- edge features (f16, 416)
__global__ void feat_k(const float* __restrict__ atoms, const int* __restrict__ Eidx,
                       const float* __restrict__ Dn, const int* __restrict__ residx,
                       const int* __restrict__ chain, const float* __restrict__ peW,
                       const float* __restrict__ peb, _Float16* __restrict__ feat) {
    size_t tid = (size_t)blockIdx.x * blockDim.x + threadIdx.x;
    if (tid >= (size_t)MEDGE * 26) return;
    int   chunk = (int)(tid % 26);
    size_t m    = tid / 26;
    int bn  = (int)(m / KNBR);
    int b   = bn / NRES, i = bn % NRES;
    int nbr = Eidx[m];
    _Float16* out = feat + m * 416 + (size_t)chunk * 16;
    if (chunk == 0) {
        int off = residx[bn] - residx[b * NRES + nbr];
        int sc  = (chain[bn] == chain[b * NRES + nbr]) ? 1 : 0;
        int d   = sc ? min(max(off + 32, 0), 64) : 65;
        #pragma unroll
        for (int c = 0; c < 16; ++c) out[c] = (_Float16)(peW[d * 16 + c] + peb[c]);
    } else {
        float dd;
        if (chunk == 1) {
            dd = Dn[m];
        } else {
            int pa = PAIR_A[chunk - 2], pb = PAIR_B[chunk - 2];
            const float* ai = atoms + (size_t)bn * 15 + pa * 3;
            const float* an = atoms + ((size_t)b * NRES + nbr) * 15 + pb * 3;
            float dx = ai[0]-an[0], dy = ai[1]-an[1], dz = ai[2]-an[2];
            dd = sqrtf(dx*dx + dy*dy + dz*dz + 1e-6f);
        }
        #pragma unroll
        for (int r = 0; r < 16; ++r) {
            float mu = 2.0f + (20.0f / 15.0f) * (float)r;
            float z  = (dd - mu) * (1.0f / 1.25f);
            out[r] = (_Float16)expf(-z * z);
        }
    }
}

// ------------------------------------------------------------- WMMA GEMM
// C[M,Nfull] = act(A[M,Kdim] @ W[Kdim,Nfull] + bias). Block: 128 rows x 128 cols,
// 256 threads = 8 waves, wave w owns rows [w*16,w*16+16). f16 math, f32 accum.
// AF16: A element type (f16/f32). CF16: output element type (f16/f32).
#define LDST 40   // half stride (80B => 16B aligned fragments, conflict padding)
template<int AF16, int CF16>
__global__ void __launch_bounds__(256) gemm128_t(const void* __restrict__ Aq,
                                                 const float* __restrict__ W,
                                                 const float* __restrict__ bias,
                                                 void* __restrict__ Cq,
                                                 int Kdim, int Nfull, int actGelu) {
    __shared__ __align__(16) _Float16 As[128 * LDST];
    __shared__ __align__(16) _Float16 Ws[128 * LDST];
    const int row0 = blockIdx.x * 128;
    const int col0 = blockIdx.y * 128;
    const int t    = threadIdx.x;
    const int wave = t >> 5, lane = t & 31;
    const int lm = lane & 15, kg = lane >> 4;
    const int ko = kg * 8;
    f8 acc[8] = {};
    for (int k0 = 0; k0 < Kdim; k0 += 32) {
        // ---- stage A tile (128 rows x 32 k) : vectorized, branch-free
        if (AF16) {
            const _Float16* Ah = (const _Float16*)Aq;
            #pragma unroll
            for (int u = 0; u < 2; ++u) {
                int idx = t * 16 + u * 8;          // 16B-aligned runs
                int r = idx >> 5, kk = idx & 31;
                *(h8*)&As[r * LDST + kk] =
                    *(const h8*)(Ah + (size_t)(row0 + r) * Kdim + k0 + kk);
            }
        } else {
            const float* Af = (const float*)Aq;
            #pragma unroll
            for (int u = 0; u < 4; ++u) {
                int idx = t * 16 + u * 4;
                int r = idx >> 5, kk = idx & 31;
                f4 v = *(const f4*)(Af + (size_t)(row0 + r) * Kdim + k0 + kk);
                h4 hv;
                hv[0] = (_Float16)v[0]; hv[1] = (_Float16)v[1];
                hv[2] = (_Float16)v[2]; hv[3] = (_Float16)v[3];
                *(h4*)&As[r * LDST + kk] = hv;
            }
        }
        // prefetch next A tile (one cacheline hint per thread)
        if (k0 + 32 < Kdim) {
            size_t ga = (size_t)(row0 + (t >> 1)) * Kdim + k0 + 32;
            __builtin_prefetch((const char*)Aq + ga * (AF16 ? 2 : 4), 0, 1);
        }
        // ---- stage W^T tile (128 cols x 32 k): consecutive lanes read
        //      consecutive columns of a W row => coalesced global loads
        #pragma unroll
        for (int u = 0; u < 16; ++u) {
            int idx = t + u * 256;                 // 0..4095
            int n = idx & 127, kk = idx >> 7;
            Ws[n * LDST + kk] = (_Float16)W[(size_t)(k0 + kk) * Nfull + col0 + n];
        }
        __syncthreads();
        // A fragment: row = wave*16+lm, K per ISA 16-bit A 16x32 layout
        const int arow = wave * 16 + lm;
        h8 alo = *(const h8*)&As[arow * LDST + ko];
        h8 ahi = *(const h8*)&As[arow * LDST + 16 + ko];
        h16 a;
        #pragma unroll
        for (int j = 0; j < 8; ++j) { a[j] = alo[j]; a[8 + j] = ahi[j]; }
        #pragma unroll
        for (int nt = 0; nt < 8; ++nt) {
            int bcol = nt * 16 + lm;
            h8 blo = *(const h8*)&Ws[bcol * LDST + ko];
            h8 bhi = *(const h8*)&Ws[bcol * LDST + 16 + ko];
            h16 bf;
            #pragma unroll
            for (int j = 0; j < 8; ++j) { bf[j] = blo[j]; bf[8 + j] = bhi[j]; }
            acc[nt] = __builtin_amdgcn_wmma_f32_16x16x32_f16(
                false, a, false, bf, (short)0, acc[nt], false, false);
        }
        __syncthreads();
    }
    const int crow0 = row0 + wave * 16 + kg * 8;   // C layout: vgpr r -> M=r (+8 hi lanes)
    #pragma unroll
    for (int nt = 0; nt < 8; ++nt) {
        int col = col0 + nt * 16 + lm;
        float bb = bias ? bias[col] : 0.f;
        #pragma unroll
        for (int vr = 0; vr < 8; ++vr) {
            float v = acc[nt][vr] + bb;
            if (actGelu) v = gelu_exact(v);
            if (CF16) ((_Float16*)Cq)[(size_t)(crow0 + vr) * Nfull + col] = (_Float16)v;
            else      ((float*)Cq)   [(size_t)(crow0 + vr) * Nfull + col] = v;
        }
    }
}

// -------------------------------------------------- layernorm over 128 cols
__global__ void __launch_bounds__(128) ln_row_k(const float* __restrict__ x,
                                                const float* __restrict__ r,
                                                const float* __restrict__ g,
                                                const float* __restrict__ bt,
                                                const float* __restrict__ mrow,
                                                float* __restrict__ out) {
    __shared__ float s[128];
    const int row = blockIdx.x, c = threadIdx.x;
    size_t base = (size_t)row * 128 + c;
    float v = x[base] + (r ? r[base] : 0.f);
    s[c] = v; __syncthreads();
    for (int st = 64; st > 0; st >>= 1) { if (c < st) s[c] += s[c + st]; __syncthreads(); }
    float mu = s[0] * (1.f / 128.f); __syncthreads();
    float d = v - mu; s[c] = d * d; __syncthreads();
    for (int st = 64; st > 0; st >>= 1) { if (c < st) s[c] += s[c + st]; __syncthreads(); }
    float var = s[0] * (1.f / 128.f);
    float y = d * rsqrtf(var + 1e-5f) * g[c] + bt[c];
    if (mrow) y *= mrow[row];
    out[base] = y;
}

// -------- node update: hV + sum_k(w*m)/30 then LN (one block per node row)
__global__ void __launch_bounds__(128) node_update_k(const float* __restrict__ hV,
                                                     const float* __restrict__ m3,
                                                     const float* __restrict__ watt,
                                                     const float* __restrict__ g,
                                                     const float* __restrict__ bt,
                                                     float* __restrict__ out) {
    __shared__ float s[128];
    const int row = blockIdx.x, c = threadIdx.x;
    size_t mb = (size_t)row * KNBR;
    float acc = 0.f;
    for (int k = 0; k < KNBR; ++k) {
        float w = watt ? watt[mb + k] : 1.f;
        acc += w * m3[(mb + k) * 128 + c];
    }
    float v = hV[(size_t)row * 128 + c] + acc * (1.f / 30.f);
    s[c] = v; __syncthreads();
    for (int st = 64; st > 0; st >>= 1) { if (c < st) s[c] += s[c + st]; __syncthreads(); }
    float mu = s[0] * (1.f / 128.f); __syncthreads();
    float d = v - mu; s[c] = d * d; __syncthreads();
    for (int st = 64; st > 0; st >>= 1) { if (c < st) s[c] += s[c + st]; __syncthreads(); }
    float var = s[0] * (1.f / 128.f);
    out[(size_t)row * 128 + c] = d * rsqrtf(var + 1e-5f) * g[c] + bt[c];
}

// ------------------------------------------------------------------ helpers
__global__ void zero_k(float* __restrict__ p, size_t n) {
    size_t i = (size_t)blockIdx.x * blockDim.x + threadIdx.x;
    if (i < n) p[i] = 0.f;
}
__global__ void embed_k(const int* __restrict__ S, const float* __restrict__ WsE,
                        float* __restrict__ hS) {
    size_t i = (size_t)blockIdx.x * blockDim.x + threadIdx.x;
    if (i >= (size_t)NV * 128) return;
    int row = (int)(i >> 7), c = (int)(i & 127);
    hS[i] = WsE[S[row] * 128 + c];
}
// fixed forward-masked encoder context: fw = mask_fw * [hE, 0, hVenc[nbr]]  (f16)
__global__ void exvfw_k(const float* __restrict__ hE, const float* __restrict__ hV,
                        const int* __restrict__ Eidx, const float* __restrict__ mask,
                        _Float16* __restrict__ fw) {
    size_t tid = (size_t)blockIdx.x * blockDim.x + threadIdx.x;
    if (tid >= (size_t)MEDGE * 384) return;
    size_t m = tid / 384; int j = (int)(tid % 384);
    int bn = (int)(m / KNBR), b = bn / NRES, i = bn % NRES;
    int nbr = Eidx[m];
    float bw  = (nbr < i) ? 1.f : 0.f;       // identity decoding order
    float mfw = mask[bn] * (1.f - bw);
    float v;
    if (j < 128)      v = hE[m * 128 + j];
    else if (j < 256) v = 0.f;
    else              v = hV[((size_t)b * NRES + nbr) * 128 + (j - 256)];
    fw[tid] = (_Float16)(mfw * v);
}
// encoder h_EV = [hV_i, hE, hV_nbr] (f16, 384)
__global__ void enc_concat_k(const float* __restrict__ hV, const float* __restrict__ hE,
                             const int* __restrict__ Eidx, _Float16* __restrict__ big) {
    size_t tid = (size_t)blockIdx.x * blockDim.x + threadIdx.x;
    if (tid >= (size_t)MEDGE * 384) return;
    size_t m = tid / 384; int j = (int)(tid % 384);
    int bn = (int)(m / KNBR), b = bn / NRES;
    float v;
    if (j < 128)      v = hV[(size_t)bn * 128 + j];
    else if (j < 256) v = hE[m * 128 + (j - 128)];
    else              v = hV[((size_t)b * NRES + Eidx[m]) * 128 + (j - 256)];
    big[tid] = (_Float16)v;
}
// decoder h_EV = [hV_i, mask_bw*[hE, hS[nbr], hV[nbr]] + fw] (f16, 512)
__global__ void dec_concat_k(const float* __restrict__ hV, const float* __restrict__ hE,
                             const float* __restrict__ hS, const _Float16* __restrict__ fw,
                             const int* __restrict__ Eidx, const float* __restrict__ mask,
                             _Float16* __restrict__ big) {
    size_t tid = (size_t)blockIdx.x * blockDim.x + threadIdx.x;
    if (tid >= (size_t)MEDGE * 512) return;
    size_t m = tid / 512; int j = (int)(tid % 512);
    int bn = (int)(m / KNBR), b = bn / NRES, i = bn % NRES;
    float val;
    if (j < 128) {
        val = hV[(size_t)bn * 128 + j];
    } else {
        int jj = j - 128;
        int nbr = Eidx[m];
        float fwd = (float)fw[m * 384 + jj];
        float bwv;
        if (jj < 128)      bwv = hE[m * 128 + jj];
        else if (jj < 256) bwv = hS[((size_t)b * NRES + nbr) * 128 + (jj - 128)];
        else               bwv = hV[((size_t)b * NRES + nbr) * 128 + (jj - 256)];
        float mbw = mask[bn] * ((nbr < i) ? 1.f : 0.f);
        val = mbw * bwv + fwd;
    }
    big[tid] = (_Float16)val;
}
// output projection 128 -> 21
__global__ void __launch_bounds__(32) out_proj_k(const float* __restrict__ hV,
                                                 const float* __restrict__ Wo,
                                                 const float* __restrict__ bo,
                                                 float* __restrict__ out) {
    const int row = blockIdx.x, c = threadIdx.x;
    if (c >= VOCABSZ) return;
    float s = 0.f;
    for (int d = 0; d < 128; ++d) s += hV[(size_t)row * 128 + d] * Wo[d * VOCABSZ + c];
    out[(size_t)row * VOCABSZ + c] = s + bo[c];
}

// =========================================================== host sequencing
extern "C" void kernel_launch(void* const* d_in, const int* in_sizes, int n_in,
                              void* d_out, int out_size, void* d_ws, size_t ws_size,
                              hipStream_t stream) {
    (void)in_sizes; (void)n_in; (void)out_size; (void)ws_size;
    // inputs (setup_inputs insertion order, params flattened depth-first)
    const float* X      = (const float*)d_in[0];
    const int*   S      = (const int*)d_in[1];
    const float* mask   = (const float*)d_in[2];
    const int*   residx = (const int*)d_in[4];
    const int*   chain  = (const int*)d_in[5];
    const float* peW    = (const float*)d_in[6];
    const float* peb    = (const float*)d_in[7];
    const float* edgeW  = (const float*)d_in[8];
    const float* lnEg   = (const float*)d_in[9];
    const float* lnEb   = (const float*)d_in[10];
    const float* WeW    = (const float*)d_in[11];
    const float* Web    = (const float*)d_in[12];
    const float* WsE    = (const float*)d_in[13];
    const float* P[52];
    for (int i = 0; i < 52; ++i) P[i] = (const float*)d_in[i];
    const float* WoutW  = (const float*)d_in[50];
    const float* Woutb  = (const float*)d_in[51];

    // workspace carve (all offsets 256B aligned)
    char* ws = (char*)d_ws;
    size_t off = 0;
    auto carve = [&](size_t bytes) -> char* {
        char* p = ws + off; off += (bytes + 255) & ~(size_t)255; return p;
    };
    float* atoms = (float*)carve((size_t)NV * 15 * 4);
    int*   Eidx  = (int*)  carve((size_t)MEDGE * 4);
    float* Dn    = (float*)carve((size_t)MEDGE * 4);
    float* matt  = (float*)carve((size_t)MEDGE * 4);
    float* hV    = (float*)carve((size_t)NV * HDIM * 4);
    float* hVt   = (float*)carve((size_t)NV * HDIM * 4);
    float* hS    = (float*)carve((size_t)NV * HDIM * 4);
    float* ffb   = (float*)carve((size_t)NV * 4 * HDIM * 4);   // f16 used, f32 sized
    float* hE    = (float*)carve((size_t)MEDGE * HDIM * 4);
    float* m1    = (float*)carve((size_t)MEDGE * HDIM * 4);    // f16 or f32 per stage
    float* m2    = (float*)carve((size_t)MEDGE * HDIM * 4);
    _Float16* fw  = (_Float16*)carve((size_t)MEDGE * 384 * 2);
    _Float16* big = (_Float16*)carve((size_t)MEDGE * 512 * 2);

    auto gemm = [&](const void* A, int aF16, const float* W, const float* bias,
                    void* C, int cF16, int Mrows, int Kd, int Ncols, int act) {
        dim3 g(Mrows / 128, Ncols / 128), blk(256);
        if (aF16) {
            if (cF16) gemm128_t<1,1><<<g, blk, 0, stream>>>(A, W, bias, C, Kd, Ncols, act);
            else      gemm128_t<1,0><<<g, blk, 0, stream>>>(A, W, bias, C, Kd, Ncols, act);
        } else {
            if (cF16) gemm128_t<0,1><<<g, blk, 0, stream>>>(A, W, bias, C, Kd, Ncols, act);
            else      gemm128_t<0,0><<<g, blk, 0, stream>>>(A, W, bias, C, Kd, Ncols, act);
        }
    };
    auto gblocks = [](size_t n) { return (unsigned)((n + 255) / 256); };

    // ---- features
    geom_k<<<gblocks(NV), 256, 0, stream>>>(X, atoms);
    topk_k<<<NV, 256, 0, stream>>>(atoms, mask, Eidx, Dn);
    maskatt_k<<<gblocks(MEDGE), 256, 0, stream>>>(mask, Eidx, matt);
    feat_k<<<gblocks((size_t)MEDGE * 26), 256, 0, stream>>>(
        atoms, Eidx, Dn, residx, chain, peW, peb, big);
    gemm(big, 1, edgeW, nullptr, m1, 0, MEDGE, 416, 128, 0);       // edge embed
    ln_row_k<<<MEDGE, 128, 0, stream>>>(m1, nullptr, lnEg, lnEb, nullptr, m2);
    gemm(m2, 0, WeW, Web, hE, 0, MEDGE, 128, 128, 0);              // h_E
    zero_k<<<gblocks((size_t)NV * HDIM), 256, 0, stream>>>(hV, (size_t)NV * HDIM);

    // ---- encoder
    for (int l = 0; l < NLAY; ++l) {
        const float *W1 = P[14]+(size_t)l*384*128, *b1 = P[15]+l*128;
        const float *W2 = P[16]+(size_t)l*128*128, *b2 = P[17]+l*128;
        const float *W3 = P[18]+(size_t)l*128*128, *b3 = P[19]+l*128;
        const float *W11= P[20]+(size_t)l*384*128, *b11= P[21]+l*128;
        const float *W12= P[22]+(size_t)l*128*128, *b12= P[23]+l*128;
        const float *W13= P[24]+(size_t)l*128*128, *b13= P[25]+l*128;
        const float *l1g= P[26]+l*128, *l1b= P[27]+l*128;
        const float *l2g= P[28]+l*128, *l2b= P[29]+l*128;
        const float *l3g= P[30]+l*128, *l3b= P[31]+l*128;
        const float *fWi= P[32]+(size_t)l*128*512, *fbi= P[33]+l*512;
        const float *fWo= P[34]+(size_t)l*512*128, *fbo= P[35]+l*128;

        enc_concat_k<<<gblocks((size_t)MEDGE*384), 256, 0, stream>>>(hV, hE, Eidx, big);
        gemm(big, 1, W1, b1, m1, 1, MEDGE, 384, 128, 1);   // f16 chain
        gemm(m1,  1, W2, b2, m2, 1, MEDGE, 128, 128, 1);
        gemm(m2,  1, W3, b3, m1, 0, MEDGE, 128, 128, 0);   // f32 for reduction
        node_update_k<<<NV, 128, 0, stream>>>(hV, m1, matt, l1g, l1b, hV);
        gemm(hV,  0, fWi, fbi, ffb, 1, NV, 128, 512, 1);
        gemm(ffb, 1, fWo, fbo, hVt, 0, NV, 512, 128, 0);
        ln_row_k<<<NV, 128, 0, stream>>>(hV, hVt, l2g, l2b, mask, hV);
        enc_concat_k<<<gblocks((size_t)MEDGE*384), 256, 0, stream>>>(hV, hE, Eidx, big);
        gemm(big, 1, W11, b11, m1, 1, MEDGE, 384, 128, 1);
        gemm(m1,  1, W12, b12, m2, 1, MEDGE, 128, 128, 1);
        gemm(m2,  1, W13, b13, m1, 0, MEDGE, 128, 128, 0);
        ln_row_k<<<MEDGE, 128, 0, stream>>>(hE, m1, l3g, l3b, nullptr, hE);
    }

    // ---- decoder prep
    embed_k<<<gblocks((size_t)NV * HDIM), 256, 0, stream>>>(S, WsE, hS);
    exvfw_k<<<gblocks((size_t)MEDGE * 384), 256, 0, stream>>>(hE, hV, Eidx, mask, fw);

    // ---- decoder
    for (int l = 0; l < NLAY; ++l) {
        const float *W1 = P[36]+(size_t)l*512*128, *b1 = P[37]+l*128;
        const float *W2 = P[38]+(size_t)l*128*128, *b2 = P[39]+l*128;
        const float *W3 = P[40]+(size_t)l*128*128, *b3 = P[41]+l*128;
        const float *l1g= P[42]+l*128, *l1b= P[43]+l*128;
        const float *l2g= P[44]+l*128, *l2b= P[45]+l*128;
        const float *fWi= P[46]+(size_t)l*128*512, *fbi= P[47]+l*512;
        const float *fWo= P[48]+(size_t)l*512*128, *fbo= P[49]+l*128;

        dec_concat_k<<<gblocks((size_t)MEDGE*512), 256, 0, stream>>>(
            hV, hE, hS, fw, Eidx, mask, big);
        gemm(big, 1, W1, b1, m1, 1, MEDGE, 512, 128, 1);
        gemm(m1,  1, W2, b2, m2, 1, MEDGE, 128, 128, 1);
        gemm(m2,  1, W3, b3, m1, 0, MEDGE, 128, 128, 0);
        node_update_k<<<NV, 128, 0, stream>>>(hV, m1, nullptr, l1g, l1b, hV);
        gemm(hV,  0, fWi, fbi, ffb, 1, NV, 128, 512, 1);
        gemm(ffb, 1, fWo, fbo, hVt, 0, NV, 512, 128, 0);
        ln_row_k<<<NV, 128, 0, stream>>>(hV, hVt, l2g, l2b, mask, hV);
    }

    // ---- logits
    out_proj_k<<<NV, 32, 0, stream>>>(hV, WoutW, Woutb, (float*)d_out);
}
